// DecisionStep_38577396253275
// MI455X (gfx1250) — matrix operand order repaired and checked by reading, not compiled, not run on hardware
//
#include <hip/hip_runtime.h>
#include <hip/hip_bf16.h>
#include <math.h>

// TabNet decision step, one workgroup per 128-row ghost-batch chunk.
// 256 threads = 8 wave32 waves; wave w owns rows 16w..16w+15 of the chunk.
// GEMMs use v_wmma_f32_16x16x32_bf16; weight k-slices are staged into a
// double-buffered LDS tile with GLOBAL_LOAD_ASYNC_TO_LDS_B128 (ASYNCcnt),
// overlapping the copy of slice i+1 with the WMMAs of slice i.

#define LDA 264   // padded bf16 row stride of activation buffer (256 + 8)
#define LDB 40    // padded bf16 row stride of staged weight k-slice (32 + 8)
#define SBSZ (256 * LDB)   // one sB buffer, in bf16 elements

typedef __bf16 v16bf __attribute__((ext_vector_type(16)));
typedef __bf16 v8bf  __attribute__((ext_vector_type(8)));
typedef float  v8f   __attribute__((ext_vector_type(8)));

static __device__ __forceinline__ v16bf make_frag(v8bf lo, v8bf hi) {
  return __builtin_shufflevector(lo, hi, 0,1,2,3,4,5,6,7,8,9,10,11,12,13,14,15);
}

// 64B per lane: four async b128 copies global -> LDS (INST_OFFSET applies to both sides).
static __device__ __forceinline__ void async_copy64(uint32_t lds_off, const void* gaddr) {
  asm volatile(
      "global_load_async_to_lds_b128 %0, %1, off\n\t"
      "global_load_async_to_lds_b128 %0, %1, off offset:16\n\t"
      "global_load_async_to_lds_b128 %0, %1, off offset:32\n\t"
      "global_load_async_to_lds_b128 %0, %1, off offset:48"
      :: "v"(lds_off), "v"(gaddr) : "memory");
}

__global__ void cvt_bf16_kernel(const float* __restrict__ src, __bf16* __restrict__ dst, int n) {
  const int i = blockIdx.x * 256 + threadIdx.x;
  if (i < n) dst[i] = (__bf16)src[i];
}

__global__ __launch_bounds__(256)
void tabnet_decision_step(
    const float* __restrict__ x,
    const float* __restrict__ a,
    const float* __restrict__ prior,
    const float* __restrict__ w_att,
    const float* __restrict__ bn_att_w, const float* __restrict__ bn_att_b,
    const float* __restrict__ w_s1,
    const float* __restrict__ bn_s1_w, const float* __restrict__ bn_s1_b,
    const float* __restrict__ w_s2,
    const float* __restrict__ bn_s2_w, const float* __restrict__ bn_s2_b,
    const float* __restrict__ w_d1,
    const float* __restrict__ bn_d1_w, const float* __restrict__ bn_d1_b,
    const float* __restrict__ w_d2,
    const float* __restrict__ bn_d2_w, const float* __restrict__ bn_d2_b,
    const __bf16* __restrict__ wb_att, const __bf16* __restrict__ wb_s1,
    const __bf16* __restrict__ wb_s2,  const __bf16* __restrict__ wb_d1,
    const __bf16* __restrict__ wb_d2,
    float* __restrict__ out, int nrows)
{
  __shared__ __align__(16) __bf16 sA[128 * LDA];   // activation buffer (bf16, A operand)
  __shared__ __align__(16) __bf16 sB[2 * SBSZ];    // double-buffered weight k-slice (bf16)
  __shared__ float sRed[2 * 16 * 256];             // GBN partial sums / sq-sums
  __shared__ float sScale[256];
  __shared__ float sShift[256];

  const int tid  = threadIdx.x;
  const int wave = tid >> 5;
  const int lane = tid & 31;
  const int half = lane >> 4;
  const int nl   = lane & 15;
  const int r0   = blockIdx.x * 128;
  const int rowL = wave * 16 + half * 8;   // first local row held by this lane in C layout

  float* out_a  = out;                                  // (N,64)
  float* out_d  = out + (size_t)nrows * 64;             // (N,64)
  float* out_np = out + (size_t)nrows * 128;            // (N,256)
  float* out_m  = out + (size_t)nrows * 384;            // (N,256)

  // 16 WMMAs of one 32-wide k-slice held in sB buffer `buf`
  auto wmma_slice = [&](int buf, int k0, v8f* acc) {
    const __bf16* abase = sA + (wave * 16 + nl) * LDA + k0 + half * 16;
    v16bf afrag = make_frag(*(const v8bf*)abase, *(const v8bf*)(abase + 8));
    const __bf16* bb = sB + buf * SBSZ;
    #pragma unroll
    for (int t = 0; t < 16; t++) {
      const __bf16* bbase = bb + (t * 16 + nl) * LDB + half * 16;
      v16bf bfrag = make_frag(*(const v8bf*)bbase, *(const v8bf*)(bbase + 8));
      acc[t] = __builtin_amdgcn_wmma_f32_16x16x32_bf16(
          false, afrag, false, bfrag, (short)0, acc[t], false, false);
    }
  };

  // ---- GEMM: C[128x256] += sA[128xK] * W^T ; W is [256 x K] row-major ----
  // Wb (bf16 copy in d_ws): async double-buffered staging; else f32 convert path.
  auto run_gemm = [&](const float* __restrict__ W, const __bf16* __restrict__ Wb,
                      int K, v8f* acc) {
    const int nk = K >> 5;
    if (Wb) {
      __syncthreads();   // all waves done reading sB from the previous GEMM
      async_copy64((uint32_t)(uintptr_t)(sB + tid * LDB), Wb + (size_t)tid * K);
      for (int i = 0; i < nk; i++) {
        const int cur = i & 1;
        __syncthreads();                       // readers of buf[1-cur] are done
        if (i + 1 < nk) {
          async_copy64((uint32_t)(uintptr_t)(sB + (1 - cur) * SBSZ + tid * LDB),
                       Wb + (size_t)tid * K + (i + 1) * 32);
          asm volatile("s_wait_asynccnt 0x4" ::: "memory");  // slice i landed (in-order)
        } else {
          asm volatile("s_wait_asynccnt 0x0" ::: "memory");
        }
        __syncthreads();                       // slice i visible to all waves
        wmma_slice(cur, i * 32, acc);
      }
    } else {
      // fallback: synchronous staging with on-the-fly f32 -> bf16 convert (buffer 0)
      for (int i = 0; i < nk; i++) {
        const int k0 = i * 32;
        __syncthreads();
        {
          const float* wrow = W + (size_t)tid * K + k0;
          __bf16* dst = sB + tid * LDB;
          #pragma unroll
          for (int j = 0; j < 32; j += 8) {
            float4 f0 = *(const float4*)(wrow + j);
            float4 f1 = *(const float4*)(wrow + j + 4);
            v8bf b;
            b[0]=(__bf16)f0.x; b[1]=(__bf16)f0.y; b[2]=(__bf16)f0.z; b[3]=(__bf16)f0.w;
            b[4]=(__bf16)f1.x; b[5]=(__bf16)f1.y; b[6]=(__bf16)f1.z; b[7]=(__bf16)f1.w;
            *(v8bf*)(dst + j) = b;
          }
        }
        __syncthreads();
        wmma_slice(0, k0, acc);
      }
    }
  };

  // ---- Ghost batch norm over the 128-row chunk (bias cancels, never added) ----
  auto run_gbn = [&](v8f* acc, const float* __restrict__ bnw, const float* __restrict__ bnb) {
    const int p = wave * 2 + half;   // 16 partials per column
    #pragma unroll
    for (int t = 0; t < 16; t++) {
      float s = 0.f, q = 0.f;
      #pragma unroll
      for (int i = 0; i < 8; i++) { float v = acc[t][i]; s += v; q += v * v; }
      const int c = t * 16 + nl;
      sRed[p * 256 + c] = s;
      sRed[4096 + p * 256 + c] = q;
    }
    __syncthreads();
    {
      const int c = tid;
      float s = 0.f, q = 0.f;
      #pragma unroll
      for (int pp = 0; pp < 16; pp++) { s += sRed[pp * 256 + c]; q += sRed[4096 + pp * 256 + c]; }
      const float mean = s * (1.0f / 128.0f);
      const float var  = fmaxf(q * (1.0f / 128.0f) - mean * mean, 0.0f);
      const float rstd = rsqrtf(var + 1e-5f);
      const float g = bnw[c] * rstd;
      sScale[c] = g;
      sShift[c] = bnb[c] - mean * g;
    }
    __syncthreads();
    #pragma unroll
    for (int t = 0; t < 16; t++) {
      const int c = t * 16 + nl;
      const float sc = sScale[c], sh = sShift[c];
      #pragma unroll
      for (int i = 0; i < 8; i++) acc[t][i] = acc[t][i] * sc + sh;
    }
  };

  // ---- GLU (+ optional residual) and store h strip to sA as next A operand ----
  auto run_glu = [&](v8f* acc, v8f* h, bool first) {
    #pragma unroll
    for (int t = 0; t < 8; t++) {
      #pragma unroll
      for (int i = 0; i < 8; i++) {
        const float u = acc[t][i], g = acc[t + 8][i];
        const float o = u / (1.0f + __expf(-g));
        h[t][i] = first ? o : 0.70710678118654752f * (h[t][i] + o);
      }
    }
    #pragma unroll
    for (int t = 0; t < 8; t++) {
      const int c = t * 16 + nl;
      #pragma unroll
      for (int i = 0; i < 8; i++)
        sA[(rowL + i) * LDA + c] = (__bf16)h[t][i];   // wave-local rows, in-order DS
    }
  };

  // ================= Attention stage =================
  {
    // stage 'a' chunk (128x64 f32) into sA as bf16
    const int row = tid >> 1;
    const int cb  = (tid & 1) * 32;
    const float* src = a + (size_t)(r0 + row) * 64 + cb;
    __bf16* dst = sA + row * LDA + cb;
    #pragma unroll
    for (int j = 0; j < 32; j += 8) {
      float4 f0 = *(const float4*)(src + j);
      float4 f1 = *(const float4*)(src + j + 4);
      v8bf b;
      b[0]=(__bf16)f0.x; b[1]=(__bf16)f0.y; b[2]=(__bf16)f0.z; b[3]=(__bf16)f0.w;
      b[4]=(__bf16)f1.x; b[5]=(__bf16)f1.y; b[6]=(__bf16)f1.z; b[7]=(__bf16)f1.w;
      *(v8bf*)(dst + j) = b;
    }
  }

  v8f acc[16];
  const v8f vzero = {0.f,0.f,0.f,0.f,0.f,0.f,0.f,0.f};
  #pragma unroll
  for (int t = 0; t < 16; t++) acc[t] = vzero;

  run_gemm(w_att, wb_att, 64, acc);
  run_gbn(acc, bn_att_w, bn_att_b);

  // The repo's ascending-sort sparsemax always selects all K elements
  // (g(r)=1+r*s[r]-cumsum[r] is non-decreasing and g(K-1)>=1), so it reduces to
  //   m = relu(z - (rowsum(z)+1)/255)
  float pr[16][8];
  #pragma unroll
  for (int t = 0; t < 16; t++) {
    const int c = t * 16 + nl;
    #pragma unroll
    for (int i = 0; i < 8; i++) {
      const size_t g = (size_t)(r0 + rowL + i) * 256 + c;
      pr[t][i] = prior[g];
      acc[t][i] *= pr[t][i];            // zp = al * prior
    }
  }
  float rs[8];
  #pragma unroll
  for (int i = 0; i < 8; i++) {
    float s = 0.f;
    #pragma unroll
    for (int t = 0; t < 16; t++) s += acc[t][i];
    rs[i] = s;
  }
  #pragma unroll
  for (int i = 0; i < 8; i++) {         // butterfly over the 16-lane column group
    rs[i] += __shfl_xor(rs[i], 1, 32);
    rs[i] += __shfl_xor(rs[i], 2, 32);
    rs[i] += __shfl_xor(rs[i], 4, 32);
    rs[i] += __shfl_xor(rs[i], 8, 32);
  }
  float tau[8];
  #pragma unroll
  for (int i = 0; i < 8; i++) tau[i] = (rs[i] + 1.0f) * (1.0f / 255.0f);

  #pragma unroll
  for (int t = 0; t < 16; t++) {
    const int c = t * 16 + nl;
    #pragma unroll
    for (int i = 0; i < 8; i++) {
      const int   rloc = rowL + i;
      const size_t g = (size_t)(r0 + rloc) * 256 + c;
      const float mm = fmaxf(acc[t][i] - tau[i], 0.0f);
      out_m[g]  = mm;
      out_np[g] = pr[t][i] * (1.5f - mm);
      sA[rloc * LDA + c] = (__bf16)(x[g] * mm);   // masked features -> next A operand
    }
  }

  // ================= GLU feature/decision chain =================
  v8f h[8];
  #pragma unroll
  for (int t = 0; t < 16; t++) acc[t] = vzero;
  run_gemm(w_s1, wb_s1, 256, acc);
  run_gbn(acc, bn_s1_w, bn_s1_b);
  run_glu(acc, h, true);

  #pragma unroll
  for (int t = 0; t < 16; t++) acc[t] = vzero;
  run_gemm(w_s2, wb_s2, 128, acc);
  run_gbn(acc, bn_s2_w, bn_s2_b);
  run_glu(acc, h, false);

  #pragma unroll
  for (int t = 0; t < 16; t++) acc[t] = vzero;
  run_gemm(w_d1, wb_d1, 128, acc);
  run_gbn(acc, bn_d1_w, bn_d1_b);
  run_glu(acc, h, false);

  #pragma unroll
  for (int t = 0; t < 16; t++) acc[t] = vzero;
  run_gemm(w_d2, wb_d2, 128, acc);
  run_gbn(acc, bn_d2_w, bn_d2_b);
  run_glu(acc, h, false);

  // ================= Outputs =================
  #pragma unroll
  for (int t = 0; t < 4; t++) {
    const int c = t * 16 + nl;
    #pragma unroll
    for (int i = 0; i < 8; i++)
      out_a[(size_t)(r0 + rowL + i) * 64 + c] = h[t][i];
  }
  #pragma unroll
  for (int t = 4; t < 8; t++) {
    const int c = t * 16 + nl - 64;
    #pragma unroll
    for (int i = 0; i < 8; i++)
      out_d[(size_t)(r0 + rowL + i) * 64 + c] = fmaxf(h[t][i], 0.0f);
  }
}

extern "C" void kernel_launch(void* const* d_in, const int* in_sizes, int n_in,
                              void* d_out, int out_size, void* d_ws, size_t ws_size,
                              hipStream_t stream) {
  (void)n_in; (void)out_size;
  const float* x      = (const float*)d_in[0];
  const float* a      = (const float*)d_in[1];
  const float* prior  = (const float*)d_in[2];
  const float* w_att  = (const float*)d_in[3];
  // biases (d_in[4,8,12,16,20]) cancel under ghost batch norm -> unused
  const float* bnaw   = (const float*)d_in[5];
  const float* bnab   = (const float*)d_in[6];
  const float* w_s1   = (const float*)d_in[7];
  const float* bns1w  = (const float*)d_in[9];
  const float* bns1b  = (const float*)d_in[10];
  const float* w_s2   = (const float*)d_in[11];
  const float* bns2w  = (const float*)d_in[13];
  const float* bns2b  = (const float*)d_in[14];
  const float* w_d1   = (const float*)d_in[15];
  const float* bnd1w  = (const float*)d_in[17];
  const float* bnd1b  = (const float*)d_in[18];
  const float* w_d2   = (const float*)d_in[19];
  const float* bnd2w  = (const float*)d_in[21];
  const float* bnd2b  = (const float*)d_in[22];

  const int nrows = in_sizes[0] / 256;      // N (x is N x 256)

  // bf16 weight copies in d_ws (w_att | w_s1 | w_s2 | w_d1 | w_d2)
  const int n_att = 256 * 64, n_s1 = 256 * 256, n_h = 256 * 128;
  const size_t ws_need = (size_t)(n_att + n_s1 + 3 * n_h) * sizeof(__bf16);
  const __bf16 *wb_att = nullptr, *wb_s1 = nullptr, *wb_s2 = nullptr,
               *wb_d1 = nullptr, *wb_d2 = nullptr;
  if (d_ws && ws_size >= ws_need) {
    __bf16* wsb = (__bf16*)d_ws;
    __bf16* p_att = wsb;
    __bf16* p_s1  = p_att + n_att;
    __bf16* p_s2  = p_s1 + n_s1;
    __bf16* p_d1  = p_s2 + n_h;
    __bf16* p_d2  = p_d1 + n_h;
    cvt_bf16_kernel<<<dim3((n_att + 255) / 256), dim3(256), 0, stream>>>(w_att, p_att, n_att);
    cvt_bf16_kernel<<<dim3((n_s1  + 255) / 256), dim3(256), 0, stream>>>(w_s1,  p_s1,  n_s1);
    cvt_bf16_kernel<<<dim3((n_h   + 255) / 256), dim3(256), 0, stream>>>(w_s2,  p_s2,  n_h);
    cvt_bf16_kernel<<<dim3((n_h   + 255) / 256), dim3(256), 0, stream>>>(w_d1,  p_d1,  n_h);
    cvt_bf16_kernel<<<dim3((n_h   + 255) / 256), dim3(256), 0, stream>>>(w_d2,  p_d2,  n_h);
    wb_att = p_att; wb_s1 = p_s1; wb_s2 = p_s2; wb_d1 = p_d1; wb_d2 = p_d2;
  }

  dim3 grid(nrows / 128), block(256);
  tabnet_decision_step<<<grid, block, 0, stream>>>(
      x, a, prior, w_att, bnaw, bnab, w_s1, bns1w, bns1b,
      w_s2, bns2w, bns2b, w_d1, bnd1w, bnd1b, w_d2, bnd2w, bnd2b,
      wb_att, wb_s1, wb_s2, wb_d1, wb_d2,
      (float*)d_out, nrows);
}
